// Postprocess_43782896615945
// MI455X (gfx1250) — compile-verified, matrix operand order checked by reading, not run on hardware
//
#include <hip/hip_runtime.h>
#include <hip/hip_bf16.h>

#define BS 16
#define GS 25200
#define NC 80
#define PS 85
#define NROWS (BS * GS)
#define NTOT (NROWS * NC)
#define KNMS 4096
#define MAXDET 100
#define CONF_T 0.5f
#define IOU_T 0.4f
#define MAXWH 1024.0f

typedef float v2f __attribute__((ext_vector_type(2)));
typedef float v8f __attribute__((ext_vector_type(8)));
typedef _Float16 v16h __attribute__((ext_vector_type(16)));
typedef unsigned u32x4 __attribute__((ext_vector_type(4)));
typedef unsigned u32x8 __attribute__((ext_vector_type(8)));

// ---------------------------------------------------------------------------
// 16x16 tile of S[m][n] = area_m + area_n via WMMA (rank-2 outer sum).
// A(16x4): col0 = area_i, col1 = 1.0 ; B(4x16): row0 = 1.0, row1 = area_j.
// ---------------------------------------------------------------------------
static __device__ inline v8f area_sum_tile(float ai, float aj, bool lo) {
#if __has_builtin(__builtin_amdgcn_wmma_f32_16x16x4_f32)
  v2f a = { lo ? ai : 0.0f, lo ? 1.0f : 0.0f };
  v2f b = { lo ? 1.0f : 0.0f, lo ? aj : 0.0f };
  v8f c = {};
  return __builtin_amdgcn_wmma_f32_16x16x4_f32(false, a, false, b, (short)0, c,
                                               false, false);
#else
  v16h a = {};
  v16h b = {};
  if (lo) {
    a[0] = (_Float16)ai; a[1] = (_Float16)1.0f;
    b[0] = (_Float16)1.0f; b[1] = (_Float16)aj;
  }
  v8f c = {};
  return __builtin_amdgcn_wmma_f32_16x16x32_f16(false, a, false, b, (short)0, c,
                                                false, false);
#endif
}

// ---------------------------------------------------------------------------
// TDM: stage a contiguous block (n8 x 8-byte elements) from global into LDS.
// D# per CDNA5 ISA: group0 = {count, lds_addr,
// global_addr, type=2}; group1 = {data_size=8B, tensor_dim0/tile_dim0/stride}.
// Issued once (call from a single wave); completion via TENSORcnt.
// ---------------------------------------------------------------------------
static __device__ inline void tdm_load_to_lds(const void* gptr, unsigned lds_off,
                                              unsigned n8) {
  unsigned long long ga = (unsigned long long)gptr;
  u32x4 g0;
  g0[0] = 1u;                                                // count=1, user D#
  g0[1] = lds_off;                                           // lds_addr
  g0[2] = (unsigned)ga;                                      // global_addr lo
  g0[3] = (unsigned)((ga >> 32) & 0x01FFFFFFu) | (2u << 30); // addr[56:32]|type
  u32x8 g1;
  g1[0] = 3u << 16;                 // wg_mask=0, data_size=3 (8 bytes)
  g1[1] = (n8 & 0xFFFFu) << 16;     // tensor_dim0[15:0] -> bits 63:48
  g1[2] = (n8 >> 16) | (1u << 16);  // tensor_dim0 hi, tensor_dim1=1
  g1[3] = (n8 & 0xFFFFu) << 16;     // tile_dim0 -> bits 127:112
  g1[4] = 1u;                       // tile_dim1=1, tile_dim2=0
  g1[5] = n8;                       // tensor_dim0_stride lo
  g1[6] = 0u;
  g1[7] = 0u;
  asm volatile("tensor_load_to_lds %0, %1" :: "s"(g0), "s"(g1) : "memory");
}

// ---------------------------------------------------------------------------
// Kernel 0: zero workspace regions that are accumulated into (mask, n_cand).
// ---------------------------------------------------------------------------
__global__ void k_init(unsigned* mask_and_keep, int* n_cand, int nwords) {
  int i = blockIdx.x * blockDim.x + threadIdx.x;
  if (i == 0) *n_cand = 0;
  for (; i < nwords; i += gridDim.x * blockDim.x) mask_and_keep[i] = 0u;
}

// ---------------------------------------------------------------------------
// Kernel 1: score = cls*obj, compact candidates (obj>0.5 && score>0.5).
// Bandwidth-bound stage (~137 MB @ 23.3 TB/s ~ 6 us).
// ---------------------------------------------------------------------------
__global__ void k_score(const float* __restrict__ p, int* n_cand,
                        float* __restrict__ sc, int* __restrict__ id) {
  int row = blockIdx.x * blockDim.x + threadIdx.x;
  if (row >= NROWS) return;
  const float* pr = p + (size_t)row * PS;
  __builtin_prefetch(pr + 8 * PS, 0, 0);  // global_prefetch_b8
  float obj = pr[4];
  if (obj <= CONF_T) return;
  for (int c = 0; c < NC; ++c) {
    float s = pr[5 + c] * obj;
    if (s > CONF_T) {
      int pos = atomicAdd(n_cand, 1);
      if (pos < KNMS) { sc[pos] = s; id[pos] = row * NC + c; }
    }
  }
}

// ---------------------------------------------------------------------------
// Kernel 2: in-LDS bitonic sort of 4096 (score desc, idx asc) == jax top_k.
// ---------------------------------------------------------------------------
__global__ __launch_bounds__(1024) void k_sort(const int* __restrict__ n_cand,
                                               float* __restrict__ sc,
                                               int* __restrict__ id) {
  __shared__ float k0[KNMS];
  __shared__ int k1[KNMS];
  int n = *n_cand;
  if (n > KNMS) n = KNMS;
  for (int i = threadIdx.x; i < KNMS; i += 1024) {
    if (i < n) { k0[i] = sc[i]; k1[i] = id[i]; }
    else       { k0[i] = -1.0f; k1[i] = 0x7FFFFFFF; }
  }
  __syncthreads();
  for (int k = 2; k <= KNMS; k <<= 1) {
    for (int j = k >> 1; j > 0; j >>= 1) {
      for (int i = threadIdx.x; i < KNMS; i += 1024) {
        int ixj = i ^ j;
        if (ixj > i) {
          bool desc = (i & k) == 0;
          float sa = k0[i], sb = k0[ixj];
          int ia = k1[i], ib = k1[ixj];
          bool lhsFirst = (sa > sb) || (sa == sb && ia < ib);
          if (desc ? !lhsFirst : lhsFirst) {
            k0[i] = sb; k0[ixj] = sa; k1[i] = ib; k1[ixj] = ia;
          }
        }
      }
      __syncthreads();
    }
  }
  for (int i = threadIdx.x; i < KNMS; i += 1024) { sc[i] = k0[i]; id[i] = k1[i]; }
}

// ---------------------------------------------------------------------------
// Kernel 3: decode boxes, apply per-(batch,class) offset, emit rows7.
// ---------------------------------------------------------------------------
__global__ void k_prep(const float* __restrict__ p, const float* __restrict__ sc,
                       const int* __restrict__ id, float4* __restrict__ nb,
                       float* __restrict__ rows7, int* __restrict__ batch) {
  int slot = blockIdx.x * blockDim.x + threadIdx.x;
  if (slot >= KNMS) return;
  float s = sc[slot];
  int idx = id[slot];
  if (idx < 0 || idx >= NTOT) {  // padding slot
    nb[slot] = make_float4(-3.0e38f, -3.0e38f, -3.0e38f, -3.0e38f);
    batch[slot] = -1;
    for (int c = 0; c < 7; ++c) rows7[slot * 7 + c] = 0.0f;
    return;
  }
  int row = idx / NC;
  int cj = idx - row * NC;
  int b = row / GS;
  const float* pr = p + (size_t)row * PS;
  float cx = pr[0], cy = pr[1], w = pr[2], h = pr[3];
  float x1 = cx - w * 0.5f, y1 = cy - h * 0.5f;
  float x2 = cx + w * 0.5f, y2 = cy + h * 0.5f;
  float off = (float)b * (NC * MAXWH) + (float)cj * MAXWH;
  nb[slot] = make_float4(x1 + off, y1 + off, x2 + off, y2 + off);
  batch[slot] = b;
  rows7[slot * 7 + 0] = x1;
  rows7[slot * 7 + 1] = y1;
  rows7[slot * 7 + 2] = x2;
  rows7[slot * 7 + 3] = y2;
  rows7[slot * 7 + 4] = s;
  rows7[slot * 7 + 5] = (float)cj;
  rows7[slot * 7 + 6] = (float)b;
}

// ---------------------------------------------------------------------------
// Kernel 4: IoU>0.4 bitmask. One block per 16-row strip (ti); the whole 64 KB
// box array is TDM-staged into LDS once, then 8 waves sweep 32 j-tiles each.
// area_m+area_n tile via V_WMMA_F32_16X16X4_F32; test inter > t*(S - inter).
// ---------------------------------------------------------------------------
__global__ __launch_bounds__(256) void k_iou(const float4* __restrict__ nb,
                                             unsigned* __restrict__ mask) {
  __shared__ float4 snb[KNMS];  // 64 KB

  if (threadIdx.x < 32) {  // wave 0 issues the TDM op (issued once per wave)
    tdm_load_to_lds(nb, (unsigned)(unsigned long long)(uintptr_t)snb,
                    KNMS * 2u /* 8-byte elements */);
    __builtin_amdgcn_s_wait_tensorcnt(0);
  }
  __syncthreads();

  unsigned lane = threadIdx.x & 31u;
  unsigned wave = threadIdx.x >> 5u;
  unsigned ti = blockIdx.x;  // 0..255
  unsigned l16 = lane & 15u;
  bool lo = lane < 16u;

  // Loop-invariant i-side rows for this wave.
  unsigned mrow = ti * 16 + (lo ? 0u : 8u);
  float4 bi0 = snb[ti * 16 + l16];
  float area_i = (bi0.z - bi0.x) * (bi0.w - bi0.y);
  float4 bi[8];
#pragma unroll
  for (int v = 0; v < 8; ++v) bi[v] = snb[mrow + v];

  for (unsigned t = 0; t < 32; ++t) {
    unsigned tj = wave * 32u + t;
    float4 bj = snb[tj * 16 + l16];
    float area_j = (bj.z - bj.x) * (bj.w - bj.y);

    v8f S = area_sum_tile(area_i, area_j, lo);

    unsigned jw = tj >> 1;            // mask word within the row
    unsigned sh = (tj & 1u) << 4;     // 0 or 16
#pragma unroll
    for (int v = 0; v < 8; ++v) {
      float ix = fminf(bi[v].z, bj.z) - fmaxf(bi[v].x, bj.x);
      float iy = fminf(bi[v].w, bj.w) - fmaxf(bi[v].y, bj.y);
      ix = fmaxf(ix, 0.0f);
      iy = fmaxf(iy, 0.0f);
      float inter = ix * iy;
      bool hit = inter > IOU_T * (S[v] - inter);
      unsigned bal = __builtin_amdgcn_ballot_w32(hit);
      if (lane == 0) {
        atomicOr(&mask[(size_t)(ti * 16 + v) * 128 + jw], (bal & 0xFFFFu) << sh);
        atomicOr(&mask[(size_t)(ti * 16 + v + 8) * 128 + jw], (bal >> 16) << sh);
      }
    }
  }
}

// ---------------------------------------------------------------------------
// Kernel 5: sequential greedy NMS scan (single workgroup, bitmap in LDS).
// ---------------------------------------------------------------------------
__global__ __launch_bounds__(128) void k_scan(const float* __restrict__ sc,
                                              const unsigned* __restrict__ mask,
                                              unsigned* __restrict__ keep_out) {
  __shared__ unsigned supp[128];
  __shared__ unsigned keepw[128];
  __shared__ unsigned validw[128];
  __shared__ int flag;
  int t = threadIdx.x;
  unsigned vb = 0u;
  for (int b = 0; b < 32; ++b)
    vb |= (sc[t * 32 + b] > CONF_T ? 1u : 0u) << b;
  validw[t] = vb;
  supp[t] = 0u;
  keepw[t] = 0u;
  __syncthreads();
  for (int i = 0; i < KNMS; ++i) {
    if (t == 0) {
      int w = i >> 5, b = i & 31;
      bool k = ((validw[w] >> b) & 1u) && !((supp[w] >> b) & 1u);
      if (k) keepw[w] |= 1u << b;
      flag = k ? 1 : 0;
    }
    __syncthreads();
    if (flag) supp[t] |= mask[(size_t)i * 128 + t];
    __syncthreads();
  }
  keep_out[t] = keepw[t];
}

// ---------------------------------------------------------------------------
// Kernel 6: zero output, then per-batch ordered gather of kept rows7.
// ---------------------------------------------------------------------------
__global__ __launch_bounds__(256) void k_gather(const unsigned* __restrict__ keepw,
                                                const int* __restrict__ batch,
                                                const float* __restrict__ rows7,
                                                float* __restrict__ out) {
  for (int i = threadIdx.x; i < BS * MAXDET * 7; i += blockDim.x) out[i] = 0.0f;
  __syncthreads();
  int b = threadIdx.x;
  if (b < BS) {
    int rank = 0;
    for (int w = 0; w < 128 && rank < MAXDET; ++w) {
      unsigned word = keepw[w];
      while (word && rank < MAXDET) {
        int bit = __ffs(word) - 1;
        word &= word - 1u;
        int i = w * 32 + bit;
        if (batch[i] == b) {
          for (int c = 0; c < 7; ++c)
            out[((size_t)b * MAXDET + rank) * 7 + c] = rows7[i * 7 + c];
          ++rank;
        }
      }
    }
  }
}

extern "C" void kernel_launch(void* const* d_in, const int* in_sizes, int n_in,
                              void* d_out, int out_size, void* d_ws, size_t ws_size,
                              hipStream_t stream) {
  const float* p = (const float*)d_in[0];
  float* out = (float*)d_out;
  char* w = (char*)d_ws;

  // workspace layout (256B-aligned slabs)
  int* n_cand     = (int*)w;                                  // 4 B
  float* sc       = (float*)(w + 1024);                       // 16 KB
  int* id         = (int*)(w + 1024 + 16384);                 // 16 KB
  float4* nb      = (float4*)(w + 1024 + 32768);              // 64 KB
  float* rows7    = (float*)(w + 1024 + 32768 + 65536);       // 112 KB
  int* batch      = (int*)(w + 1024 + 32768 + 65536 + 114688);// 16 KB
  unsigned* mask  = (unsigned*)(w + 1024 + 32768 + 65536 + 114688 + 16384); // 2 MB
  unsigned* keepw = mask + (size_t)KNMS * 128;                // 512 B

  const int mask_words = KNMS * 128 + 128;

  k_init<<<1024, 256, 0, stream>>>(mask, n_cand, mask_words);
  k_score<<<(NROWS + 255) / 256, 256, 0, stream>>>(p, n_cand, sc, id);
  k_sort<<<1, 1024, 0, stream>>>(n_cand, sc, id);
  k_prep<<<KNMS / 256, 256, 0, stream>>>(p, sc, id, nb, rows7, batch);
  k_iou<<<256, 256, 0, stream>>>(nb, mask);
  k_scan<<<1, 128, 0, stream>>>(sc, mask, keepw);
  k_gather<<<1, 256, 0, stream>>>(keepw, batch, rows7, out);
}
